// SecondGeometry_88725434401337
// MI455X (gfx1250) — matrix operand order repaired and checked by reading, not compiled, not run on hardware
//
#include <hip/hip_runtime.h>
#include <math.h>

typedef __attribute__((ext_vector_type(16))) _Float16 v16h;
typedef __attribute__((ext_vector_type(8)))  _Float16 v8h;
typedef __attribute__((ext_vector_type(2)))  _Float16 v2h;
typedef __attribute__((ext_vector_type(8)))  float    v8f;

#define HID 256
#define NW  4                 // waves per workgroup
#define NTHR (NW*32)          // 128 threads
#define WGSAMP (NW*16)        // 64 samples per workgroup
#define NB 65536

// ---- fragment helpers (CDNA5 wave32 WMMA 16x16x32 f16 layouts) ----
__device__ __forceinline__ v16h load_a_frag(const _Float16* tile, int lane, int kbase) {
  const int m = lane & 15, hh = lane >> 4;
  const _Float16* p = tile + m * HID + kbase + 16 * hh;
  return *(const v16h*)p;
}
__device__ __forceinline__ v16h load_b_frag(const _Float16* slab16, int lane, int kbase) {
  const int n = lane & 15, kh = lane >> 4;
  return *(const v16h*)(slab16 + n * HID + kbase + 16 * kh);
}
__device__ __forceinline__ v8f wmma_f16(v16h a, v16h b, v8f c) {
  return __builtin_amdgcn_wmma_f32_16x16x32_f16(false, a, false, b, (short)0, c, false, false);
}
// clamped fast tanh: 2 TRANS + ~6 VALU (vs long __ocml_tanh_f32 sequence)
__device__ __forceinline__ float fast_tanh(float x) {
  x = fminf(10.0f, fmaxf(-10.0f, x));
  const float e = __expf(2.0f * x);
  return (e - 1.0f) * __builtin_amdgcn_rcpf(e + 1.0f);
}

// ---- prelude: W2 (f32) -> f16 row-major + f16 transposed copies in workspace
__global__ void SecondGeometry_cvt_w2(const float* __restrict__ W2,
                                      _Float16* __restrict__ w2f,
                                      _Float16* __restrict__ w2t) {
  const int i = blockIdx.x, j = threadIdx.x;
  const float v = W2[i * HID + j];
  const _Float16 h = (_Float16)v;
  w2f[i * HID + j] = h;   // [k][n] = W2[k][n]
  w2t[j * HID + i] = h;   // [n][k] = W2[k][n]
}

__global__ __launch_bounds__(NTHR)
void SecondGeometry_main(const float* __restrict__ x,
                         const float* __restrict__ W1,
                         const float* __restrict__ b1,
                         const float* __restrict__ b2,
                         const float* __restrict__ W3,
                         const float* __restrict__ attr,
                         const float* __restrict__ kstiff,
                         const float* __restrict__ kdiss,
                         const _Float16* __restrict__ w2f,
                         const _Float16* __restrict__ w2t,
                         float* __restrict__ out) {
  __shared__ alignas(32) _Float16 sW1T[HID * 8];        // [n][k] = W1[k][n]   (4 KB)
  __shared__ alignas(32) _Float16 sW1R[16 * HID];       // rows 0..7 = W1[q][:], 8..15 = 0 (8 KB)
  __shared__ alignas(32) _Float16 slab[16 * HID];       // shared W2 slice (8 KB)
  __shared__ alignas(32) _Float16 hT[NW][16 * HID];     // h1, later u1 (32 KB)
  __shared__ alignas(32) _Float16 zT[NW][16 * HID];     // zdot1 = W1^T v (32 KB)
  __shared__ alignas(32) _Float16 uT[NW][16 * HID];     // u2 (32 KB)
  __shared__ alignas(32) _Float16 wT[NW][16 * HID];     // udot2 (32 KB)
  __shared__ float sB1[HID], sB2[HID], sW3[HID], sAttr[8];
  __shared__ float sG[NW][16][8];
  __shared__ float sS[NW][16];

  const int tid  = threadIdx.x;
  const int lane = tid & 31;
  const int w    = tid >> 5;
  const int sbase = blockIdx.x * WGSAMP + w * 16;

  // ---- stage small weights ----
  for (int i = tid; i < HID; i += NTHR) { sB1[i] = b1[i]; sB2[i] = b2[i]; sW3[i] = W3[i]; }
  for (int i = tid; i < HID * 8; i += NTHR) {
    const int n = i >> 3, k = i & 7;
    sW1T[n * 8 + k] = (_Float16)W1[k * HID + n];
  }
  for (int i = tid; i < 16 * HID; i += NTHR) {
    const int q = i >> 8, k = i & 255;
    sW1R[q * HID + k] = (q < 8) ? (_Float16)W1[q * HID + k] : (_Float16)0.0f;
  }
  if (tid < 8) sAttr[tid] = attr[tid];
  if (tid < NW * 16) sS[tid >> 4][tid & 15] = 0.0f;
  __syncthreads();

  // ---- layer 1 (K=8, VALU): z1 = p@W1+b1 -> h1=tanh ; zdot1 = v@W1 ----
  {
    const int m = lane & 15, nh = lane >> 4;
    const float* xr = x + (size_t)(sbase + m) * 16;
    float pp[8], vv[8];
#pragma unroll
    for (int k = 0; k < 8; k++) { pp[k] = xr[k]; vv[k] = xr[8 + k]; }
    for (int n0 = 0; n0 < 128; n0 += 2) {
      const int n = nh * 128 + n0;
      const v8h wa = *(const v8h*)&sW1T[n * 8];
      const v8h wb = *(const v8h*)&sW1T[(n + 1) * 8];
      float z1a = sB1[n], zva = 0.0f, z1b = sB1[n + 1], zvb = 0.0f;
#pragma unroll
      for (int k = 0; k < 8; k++) {
        const float wka = (float)wa[k], wkb = (float)wb[k];
        z1a += pp[k] * wka; zva += vv[k] * wka;
        z1b += pp[k] * wkb; zvb += vv[k] * wkb;
      }
      v2h hp, zp;
      hp[0] = (_Float16)fast_tanh(z1a); hp[1] = (_Float16)fast_tanh(z1b);
      zp[0] = (_Float16)zva;            zp[1] = (_Float16)zvb;
      *(v2h*)&hT[w][m * HID + n] = hp;
      *(v2h*)&zT[w][m * HID + n] = zp;
    }
  }
  __syncthreads();

  // ---- hoist A fragments for layer-2 fwd+tangent (reused across all 16 nt) ----
  v16h ah[8], aq[8];
#pragma unroll
  for (int kt = 0; kt < 8; ++kt) {
    const int kb = kt * 32;
    ah[kt] = load_a_frag(hT[w], lane, kb);
    const v16h az = load_a_frag(zT[w], lane, kb);
    v16h q;
#pragma unroll
    for (int i = 0; i < 16; i++) { const _Float16 h = ah[kt][i]; q[i] = ((_Float16)1.0f - h * h) * az[i]; }
    aq[kt] = q;
  }

  // register prefetch of first W2^T slice
  v16h pre0, pre1;
  { const _Float16* src = w2t;
    pre0 = *(const v16h*)&src[tid * 32];
    pre1 = *(const v16h*)&src[tid * 32 + 16]; }

  // ---- layer 2 forward + tangent: z2 = h1@W2, t2 = (d1*zdot1)@W2 ----
  for (int nt = 0; nt < 16; ++nt) {
    __syncthreads();
    *(v16h*)&slab[tid * 32]      = pre0;
    *(v16h*)&slab[tid * 32 + 16] = pre1;
    __syncthreads();
    if (nt < 15) {                                 // prefetch next slice (latency hidden by WMMAs)
      const _Float16* src = w2t + (nt + 1) * 16 * HID;
      pre0 = *(const v16h*)&src[tid * 32];
      pre1 = *(const v16h*)&src[tid * 32 + 16];
    }
    v8f z2 = {}; v8f t2 = {};
#pragma unroll
    for (int kt = 0; kt < 8; ++kt) {
      const v16h bfr = load_b_frag(slab, lane, kt * 32);
      z2 = wmma_f16(ah[kt], bfr, z2);
      t2 = wmma_f16(aq[kt], bfr, t2);
    }
    const int n = nt * 16 + (lane & 15);
    const int mb = (lane >> 4) * 8;
    const float w3 = sW3[n], bb = sB2[n];
#pragma unroll
    for (int r = 0; r < 8; r++) {
      const float h2 = fast_tanh(z2[r] + bb);
      const float d2 = 1.0f - h2 * h2;
      uT[w][(mb + r) * HID + n] = (_Float16)(d2 * w3);
      wT[w][(mb + r) * HID + n] = (_Float16)(-2.0f * h2 * d2 * t2[r] * w3);
    }
  }

  // prefetch first W2 slice for backward phase
  { const _Float16* src = w2f;
    pre0 = *(const v16h*)&src[tid * 32];
    pre1 = *(const v16h*)&src[tid * 32 + 16]; }

  // ---- hoist A fragments for backward (u2, udot2) ----
  v16h au[8], aw[8];
#pragma unroll
  for (int kt = 0; kt < 8; ++kt) {
    au[kt] = load_a_frag(uT[w], lane, kt * 32);
    aw[kt] = load_a_frag(wT[w], lane, kt * 32);
  }

  // ---- backward: c2 = u2@W2^T, cd2 = udot2@W2^T ; build u1 (into hT), s ----
  float sacc[8] = {0, 0, 0, 0, 0, 0, 0, 0};
  for (int nt = 0; nt < 16; ++nt) {
    __syncthreads();
    *(v16h*)&slab[tid * 32]      = pre0;
    *(v16h*)&slab[tid * 32 + 16] = pre1;
    __syncthreads();
    if (nt < 15) {
      const _Float16* src = w2f + (nt + 1) * 16 * HID;
      pre0 = *(const v16h*)&src[tid * 32];
      pre1 = *(const v16h*)&src[tid * 32 + 16];
    }
    v8f c2 = {}; v8f cd2 = {};
#pragma unroll
    for (int kt = 0; kt < 8; ++kt) {
      const v16h bfr = load_b_frag(slab, lane, kt * 32);
      c2  = wmma_f16(au[kt], bfr, c2);
      cd2 = wmma_f16(aw[kt], bfr, cd2);
    }
    const int j = nt * 16 + (lane & 15);
    const int mb = (lane >> 4) * 8;
#pragma unroll
    for (int r = 0; r < 8; r++) {
      const int m = mb + r;
      const float h  = (float)hT[w][m * HID + j];
      const float zv = (float)zT[w][m * HID + j];
      const float d1 = 1.0f - h * h;
      const float c = c2[r], cd = cd2[r];
      const float u1  = d1 * c;
      const float ud1 = d1 * cd - 2.0f * h * (d1 * zv) * c;
      sacc[r] += zv * ud1;                    // s = zdot1 . udot1
      hT[w][m * HID + j] = (_Float16)u1;      // reuse hT tile as u1
    }
  }
  { const int mb = (lane >> 4) * 8;
#pragma unroll
    for (int r = 0; r < 8; r++) atomicAdd(&sS[w][mb + r], sacc[r]);
  }

  // ---- g = u1 @ W1^T (N padded 8->16) ----
  v8f gac = {};
#pragma unroll
  for (int kt = 0; kt < 8; ++kt) {
    const v16h bfr = load_b_frag(sW1R, lane, kt * 32);
    const v16h a1  = load_a_frag(hT[w], lane, kt * 32);
    gac = wmma_f16(a1, bfr, gac);
  }
  { const int q = lane & 15, mb = (lane >> 4) * 8;
    if (q < 8) {
#pragma unroll
      for (int r = 0; r < 8; r++) sG[w][mb + r][q] = gac[r];
    }
  }
  __syncthreads();

  // ---- final: acc_q = f_q - g_q*(g.f + vHv)/(1+|g|^2) ----
  if (lane < 16) {
    const int m = lane;
    const int samp = sbase + m;
    const float* xr = x + (size_t)samp * 16;
    const float ks = kstiff[0], kd = kdiss[0];
    float g[8], f[8], gg = 0.0f, gf = 0.0f;
#pragma unroll
    for (int q = 0; q < 8; q++) {
      const float gq = sG[w][m][q];
      const float fq = -(kd * xr[8 + q] + ks * (xr[q] - sAttr[q]));
      g[q] = gq; f[q] = fq; gg += gq * gq; gf += gq * fq;
    }
    const float alpha = (gf + sS[w][m]) / (1.0f + gg);
    float* o = out + (size_t)samp * 8;
#pragma unroll
    for (int q = 0; q < 8; q++) o[q] = f[q] - g[q] * alpha;
  }
}

extern "C" void kernel_launch(void* const* d_in, const int* in_sizes, int n_in,
                              void* d_out, int out_size, void* d_ws, size_t ws_size,
                              hipStream_t stream) {
  const float* x    = (const float*)d_in[0];
  const float* W1   = (const float*)d_in[1];
  const float* b1   = (const float*)d_in[2];
  const float* W2   = (const float*)d_in[3];
  const float* b2   = (const float*)d_in[4];
  const float* W3   = (const float*)d_in[5];
  const float* attr = (const float*)d_in[7];
  const float* ks   = (const float*)d_in[8];
  const float* kd   = (const float*)d_in[9];
  float* out = (float*)d_out;

  _Float16* w2f = (_Float16*)d_ws;          // 128 KB
  _Float16* w2t = w2f + HID * HID;          // 128 KB

  SecondGeometry_cvt_w2<<<dim3(HID), dim3(HID), 0, stream>>>(W2, w2f, w2t);
  SecondGeometry_main<<<dim3(NB / WGSAMP), dim3(NTHR), 0, stream>>>(
      x, W1, b1, b2, W3, attr, ks, kd, w2f, w2t, out);
}
